// MultiHeadDecoder_781684048388
// MI455X (gfx1250) — matrix operand order repaired and checked by reading, not compiled, not run on hardware
//
#include <hip/hip_runtime.h>
#include <hip/hip_bf16.h>

#define IN_F   256
#define HID    512
#define OUT_F  2048
#define N_HEADS 8
#define BATCH  4096
// heads padded to 32-row tiles: 4096 + 8*31 = 4344 -> round to 32 -> 4352
#define MPAD   4352   // 136 tiles of 32

typedef __attribute__((ext_vector_type(16))) __bf16 v16bf;
typedef __attribute__((ext_vector_type(8)))  float  v8f;
typedef __attribute__((ext_vector_type(4)))  int    v4i;
typedef __attribute__((ext_vector_type(4)))  float  v4f;

union Frag { v16bf v; v4i q[2]; };

__device__ __forceinline__ __bf16 to_bf16(float f) { return (__bf16)f; }

__device__ __forceinline__ v8f wmma_bf16(const Frag& a, const Frag& b, v8f c) {
    return __builtin_amdgcn_wmma_f32_16x16x32_bf16(false, a.v, false, b.v,
                                                   (short)0, c, false, false);
}

// ---------------- bucketing ----------------
__global__ void k_init(int* counts, int* cursor, int* perm) {
    int i = blockIdx.x * blockDim.x + threadIdx.x;
    if (i < N_HEADS) { counts[i] = 0; cursor[i] = 0; }
    if (i < MPAD) perm[i] = -1;
}

__global__ void k_count(const int* __restrict__ idx, int* counts) {
    int i = blockIdx.x * blockDim.x + threadIdx.x;
    if (i < BATCH) atomicAdd(&counts[idx[i]], 1);
}

__global__ void k_prefix(const int* __restrict__ counts, int* offs) {
    if (threadIdx.x == 0) {
        int a = 0; offs[0] = 0;
        for (int h = 0; h < N_HEADS; ++h) {
            a += (counts[h] + 31) & ~31;   // pad each head to 32-row tiles
            offs[h + 1] = a;
        }
    }
}

__global__ void k_scatter(const int* __restrict__ idx, const int* __restrict__ offs,
                          int* cursor, int* perm) {
    int i = blockIdx.x * blockDim.x + threadIdx.x;
    if (i < BATCH) {
        int h = idx[i];
        int pos = offs[h] + atomicAdd(&cursor[h], 1);
        perm[pos] = i;
    }
}

// gather X rows into permuted/padded bf16 activation buffer
__global__ void k_gatherX(const float* __restrict__ X, const int* __restrict__ perm,
                          __bf16* __restrict__ Xg) {
    int t = blockIdx.x * blockDim.x + threadIdx.x;     // MPAD * IN_F/4 threads
    int p = t / (IN_F / 4);
    int c = (t % (IN_F / 4)) * 4;
    if (p >= MPAD) return;
    int row = perm[p];
    v4f v = {};
    if (row >= 0) v = *(const v4f*)(X + (size_t)row * IN_F + c);
    __bf16* o = Xg + (size_t)p * IN_F + c;
    o[0] = to_bf16(v[0]); o[1] = to_bf16(v[1]);
    o[2] = to_bf16(v[2]); o[3] = to_bf16(v[3]);
}

// ---------------- weight transpose + fp32->bf16 ----------------
// W[h][k][n] (fp32) -> WT[h][n][k] (bf16), LDS tiled so both sides coalesce
template<int K, int N>
__global__ void k_transpose(const float* __restrict__ W, __bf16* __restrict__ WT) {
    __shared__ float tile[32][33];
    const int h  = blockIdx.z;
    const int kb = blockIdx.x * 32, nb = blockIdx.y * 32;
    const int tx = threadIdx.x, ty = threadIdx.y;
    tile[ty][tx] = W[(size_t)h * K * N + (size_t)(kb + ty) * N + (nb + tx)];
    __syncthreads();
    WT[(size_t)h * N * K + (size_t)(nb + ty) * K + (kb + tx)] = to_bf16(tile[tx][ty]);
}

// ---------------- WMMA GEMMs ----------------
// Per wave: 32(M)x32(N) output tile = 4 f32 accumulators (2 A frags x 2 B frags)
//   -> 8x16B fragment loads feed 4 WMMAs (2 loads/WMMA, 16 FLOP/byte from L0/L2)
// A frag (16x32 bf16): lane(0-15)=row M=lane; K halves {kg*8..+7, 16+kg*8..+7}, kg=lane>>4
// B frag from WT[N][K] row-major: lane N=lane&15, same K halves as A {kg*8..+7, 16+kg*8..+7}

__global__ __launch_bounds__(256)
void k_gemm1(const __bf16* __restrict__ A, const __bf16* __restrict__ WT,
             const float* __restrict__ bias, const int* __restrict__ offs,
             __bf16* __restrict__ H) {
    __shared__ int s_off[N_HEADS + 1];
    if (threadIdx.x <= N_HEADS) s_off[threadIdx.x] = offs[threadIdx.x];
    __syncthreads();
    const int mbase = blockIdx.x * 32;
    if (mbase >= s_off[N_HEADS]) return;
    int head = 0;
    while (mbase >= s_off[head + 1]) head++;

    const int lane = threadIdx.x & 31;
    const int wave = threadIdx.x >> 5;
    const int nbase = blockIdx.y * 256 + wave * 32;
    const int lrow = lane & 15;
    const int kg   = lane >> 4;

    const __bf16* ar0 = A + (size_t)(mbase + lrow) * IN_F;   // rows mbase..+15
    const __bf16* ar1 = ar0 + (size_t)16 * IN_F;             // rows mbase+16..+31
    const __bf16* bp0 = WT + ((size_t)head * HID + nbase + lrow) * IN_F + kg * 8;
    const __bf16* bp1 = bp0 + (size_t)16 * IN_F;

    v8f acc00 = {}, acc01 = {}, acc10 = {}, acc11 = {};
#pragma unroll
    for (int k0 = 0; k0 < IN_F; k0 += 32) {
        Frag a0, a1, f0, f1;
        a0.q[0] = *(const v4i*)(ar0 + k0 + kg * 8);
        a0.q[1] = *(const v4i*)(ar0 + k0 + 16 + kg * 8);
        a1.q[0] = *(const v4i*)(ar1 + k0 + kg * 8);
        a1.q[1] = *(const v4i*)(ar1 + k0 + 16 + kg * 8);
        f0.q[0] = *(const v4i*)(bp0 + k0);
        f0.q[1] = *(const v4i*)(bp0 + k0 + 16);
        f1.q[0] = *(const v4i*)(bp1 + k0);
        f1.q[1] = *(const v4i*)(bp1 + k0 + 16);
        acc00 = wmma_bf16(a0, f0, acc00);
        acc01 = wmma_bf16(a0, f1, acc01);
        acc10 = wmma_bf16(a1, f0, acc10);
        acc11 = wmma_bf16(a1, f1, acc11);
    }
    // epilogue: bias + relu -> bf16
    const int ncol = lane & 15;
    const int mhi  = (lane >> 4) * 8;
    const float bv0 = bias[head * HID + nbase + ncol];
    const float bv1 = bias[head * HID + nbase + 16 + ncol];
#pragma unroll
    for (int r = 0; r < 8; ++r) {
        const int m0 = mbase + mhi + r;
        const int m1 = m0 + 16;
        float v00 = acc00[r] + bv0; v00 = v00 > 0.f ? v00 : 0.f;
        float v01 = acc01[r] + bv1; v01 = v01 > 0.f ? v01 : 0.f;
        float v10 = acc10[r] + bv0; v10 = v10 > 0.f ? v10 : 0.f;
        float v11 = acc11[r] + bv1; v11 = v11 > 0.f ? v11 : 0.f;
        H[(size_t)m0 * HID + nbase + ncol]      = to_bf16(v00);
        H[(size_t)m0 * HID + nbase + 16 + ncol] = to_bf16(v01);
        H[(size_t)m1 * HID + nbase + ncol]      = to_bf16(v10);
        H[(size_t)m1 * HID + nbase + 16 + ncol] = to_bf16(v11);
    }
}

__global__ __launch_bounds__(256)
void k_gemm2(const __bf16* __restrict__ Hg, const __bf16* __restrict__ WT,
             const float* __restrict__ bias, const int* __restrict__ offs,
             const int* __restrict__ perm, float* __restrict__ out) {
    __shared__ int s_off[N_HEADS + 1];
    if (threadIdx.x <= N_HEADS) s_off[threadIdx.x] = offs[threadIdx.x];
    __syncthreads();
    const int mbase = blockIdx.x * 32;
    if (mbase >= s_off[N_HEADS]) return;
    int head = 0;
    while (mbase >= s_off[head + 1]) head++;

    const int lane = threadIdx.x & 31;
    const int wave = threadIdx.x >> 5;
    const int nbase = blockIdx.y * 256 + wave * 32;
    const int lrow = lane & 15;
    const int kg   = lane >> 4;

    const __bf16* ar0 = Hg + (size_t)(mbase + lrow) * HID;
    const __bf16* ar1 = ar0 + (size_t)16 * HID;
    const __bf16* bp0 = WT + ((size_t)head * OUT_F + nbase + lrow) * HID + kg * 8;
    const __bf16* bp1 = bp0 + (size_t)16 * HID;

    v8f acc00 = {}, acc01 = {}, acc10 = {}, acc11 = {};
#pragma unroll
    for (int k0 = 0; k0 < HID; k0 += 32) {
        Frag a0, a1, f0, f1;
        a0.q[0] = *(const v4i*)(ar0 + k0 + kg * 8);
        a0.q[1] = *(const v4i*)(ar0 + k0 + 16 + kg * 8);
        a1.q[0] = *(const v4i*)(ar1 + k0 + kg * 8);
        a1.q[1] = *(const v4i*)(ar1 + k0 + 16 + kg * 8);
        f0.q[0] = *(const v4i*)(bp0 + k0);
        f0.q[1] = *(const v4i*)(bp0 + k0 + 16);
        f1.q[0] = *(const v4i*)(bp1 + k0);
        f1.q[1] = *(const v4i*)(bp1 + k0 + 16);
        acc00 = wmma_bf16(a0, f0, acc00);
        acc01 = wmma_bf16(a0, f1, acc01);
        acc10 = wmma_bf16(a1, f0, acc10);
        acc11 = wmma_bf16(a1, f1, acc11);
    }
    // epilogue: bias, scatter back through permutation (fp32 out)
    const int ncol = lane & 15;
    const int mhi  = (lane >> 4) * 8;
    const float bv0 = bias[head * OUT_F + nbase + ncol];
    const float bv1 = bias[head * OUT_F + nbase + 16 + ncol];
#pragma unroll
    for (int r = 0; r < 8; ++r) {
        const int m0 = mbase + mhi + r;
        const int m1 = m0 + 16;
        const int o0 = perm[m0];
        const int o1 = perm[m1];
        if (o0 >= 0) {
            out[(size_t)o0 * OUT_F + nbase + ncol]      = acc00[r] + bv0;
            out[(size_t)o0 * OUT_F + nbase + 16 + ncol] = acc01[r] + bv1;
        }
        if (o1 >= 0) {
            out[(size_t)o1 * OUT_F + nbase + ncol]      = acc10[r] + bv0;
            out[(size_t)o1 * OUT_F + nbase + 16 + ncol] = acc11[r] + bv1;
        }
    }
}

// ---------------- launcher ----------------
extern "C" void kernel_launch(void* const* d_in, const int* in_sizes, int n_in,
                              void* d_out, int out_size, void* d_ws, size_t ws_size,
                              hipStream_t stream) {
    const float* X   = (const float*)d_in[0];
    const int*   idx = (const int*)d_in[1];
    const float* W1  = (const float*)d_in[2];
    const float* b1  = (const float*)d_in[3];
    const float* W2  = (const float*)d_in[4];
    const float* b2  = (const float*)d_in[5];
    float* out = (float*)d_out;

    // workspace carve-up (~25.7 MB)
    int* counts = (int*)d_ws;                 // 8
    int* cursor = counts + 8;                 // 8
    int* offs   = cursor + 8;                 // 9 (slot of 16)
    int* perm   = offs + 16;                  // MPAD
    __bf16* Xg  = (__bf16*)(perm + MPAD);               // [MPAD][IN_F]
    __bf16* Hg  = Xg  + (size_t)MPAD * IN_F;            // [MPAD][HID]
    __bf16* W1T = Hg  + (size_t)MPAD * HID;             // [H][HID][IN_F]
    __bf16* W2T = W1T + (size_t)N_HEADS * HID * IN_F;   // [H][OUT_F][HID]

    k_init   <<<(MPAD + 255) / 256, 256, 0, stream>>>(counts, cursor, perm);
    k_count  <<<(BATCH + 255) / 256, 256, 0, stream>>>(idx, counts);
    k_prefix <<<1, 32, 0, stream>>>(counts, offs);
    k_scatter<<<(BATCH + 255) / 256, 256, 0, stream>>>(idx, offs, cursor, perm);
    k_gatherX<<<(MPAD * (IN_F / 4) + 255) / 256, 256, 0, stream>>>(X, perm, Xg);

    dim3 tb(32, 32);
    k_transpose<IN_F, HID>  <<<dim3(IN_F / 32, HID / 32,  N_HEADS), tb, 0, stream>>>(W1, W1T);
    k_transpose<HID, OUT_F> <<<dim3(HID / 32,  OUT_F / 32, N_HEADS), tb, 0, stream>>>(W2, W2T);

    k_gemm1<<<dim3(MPAD / 32, HID / 256),   256, 0, stream>>>(Xg, W1T, b1, offs, Hg);
    k_gemm2<<<dim3(MPAD / 32, OUT_F / 256), 256, 0, stream>>>(Hg, W2T, b2, offs, perm, out);
}
